// DotProductAttention_34041910788254
// MI455X (gfx1250) — matrix operand order repaired and checked by reading, not compiled, not run on hardware
//
#include <hip/hip_runtime.h>

// -----------------------------------------------------------------------------
// Causal attention with input projections for MI455X (gfx1250, wave32, WMMA).
//
//   1) proj_gemm_f16 x3 : q1/k1/v1 = (f32 X) @ (f32 W) -> f16   [WMMA f16]
//   2) scores_gemm      : Sc = scale * q1 @ k1^T, causal mask   [WMMA f16]
//   3) softmax_rows     : P  = softmax(Sc) -> f16               [v_exp_f32]
//   4) pv_gemm          : O  = P @ v1 -> f32, K truncated at diagonal
//
// B tiles are staged in LDS *transposed* (Bt[n][k]) so every WMMA fragment
// load is 2x ds_load_b128 (contiguous per lane) instead of 16x ds_load_u16.
// -----------------------------------------------------------------------------

typedef __attribute__((ext_vector_type(16))) _Float16 v16h;
typedef __attribute__((ext_vector_type(8)))  _Float16 h8;
typedef __attribute__((ext_vector_type(2)))  _Float16 h2;
typedef __attribute__((ext_vector_type(8)))  float    v8f;

#define BATCH 4
#define SEQ   2048
#define DIM   1024
#define TM    128          // block tile M
#define TN    128          // block tile N
#define TK    32           // K step (== WMMA K)
#define LDA   (TK + 8)     // A tile row stride (f16), 80B: keeps b128 aligned
#define LDBT  (TK + 8)     // Bt tile row stride (f16)
#define NEGINF (-3.0e38f)
#define SCALE  (0.03125f)  // rsqrt(1024)

static __device__ __forceinline__ v8f wmma_f16(v16h a, v16h b, v8f c) {
  // (neg_a, A, neg_b, B, c_mod, C, reuse_a, reuse_b)
  return __builtin_amdgcn_wmma_f32_16x16x32_f16(false, a, false, b,
                                                (short)0, c, false, false);
}

// A fragment (16x32 f16): lane l<16 -> row m0+l, K {0..15};
// lanes 16-31 -> K {16..31}. Two contiguous 16B runs -> 2x ds_load_b128.
static __device__ __forceinline__ v16h load_a_frag(const _Float16* As, int m0, int lane) {
  v16h a;
  const _Float16* p = As + (m0 + (lane & 15)) * LDA + (lane >> 4) * 16;
#pragma unroll
  for (int t = 0; t < 8; ++t) {
    a[t]     = p[t];
    a[8 + t] = p[8 + t];
  }
  return a;
}

// B fragment (32x16 f16) from transposed LDS tile Bt[n][k]:
// lane reads col n0+(lane&15), K = (lane>>4)*16 .. +15 -> 32 contiguous bytes.
static __device__ __forceinline__ v16h load_b_frag_t(const _Float16* Bt, int n0, int lane) {
  v16h b;
  const _Float16* p = Bt + (n0 + (lane & 15)) * LDBT + (lane >> 4) * 16;
#pragma unroll
  for (int t = 0; t < 16; ++t) b[t] = p[t];
  return b;
}

// -------------------------- 1) projection GEMM ------------------------------
// Y(f16, MxN) = X(f32, MxK) @ W(f32, KxN); f32 -> f16 conversion while staging.
__global__ __launch_bounds__(256) void proj_gemm_f16(
    const float* __restrict__ X, const float* __restrict__ W,
    _Float16* __restrict__ Y, int M, int N, int K) {
  __shared__ alignas(16) _Float16 As[TM][LDA];
  __shared__ alignas(16) _Float16 Bt[TN][LDBT];
  const int tid  = threadIdx.x;
  const int lane = tid & 31;
  const int wave = tid >> 5;
  const int m_blk = blockIdx.x * TM;
  const int n_blk = blockIdx.y * TN;
  const int wm = (wave >> 1) * 32;   // 4x2 wave grid -> 32x64 per wave
  const int wn = (wave & 1) * 64;
  const int arow = tid >> 1, acb = (tid & 1) * 16;   // A: 16 f32 / thread
  const int bk = (tid & 15) * 2;                     // B: rows k, k+1
  const int bn = (tid >> 4) * 8;                     //    8 cols each
  v8f acc[2][4] = {};

  for (int k0 = 0; k0 < K; k0 += TK) {
    // ---- stage A (128x32 f32 -> f16) ----
    const float* xp = X + (size_t)(m_blk + arow) * K + k0 + acb;
#pragma unroll
    for (int t = 0; t < 4; ++t) {
      float4 f = reinterpret_cast<const float4*>(xp)[t];
      As[arow][acb + 4 * t + 0] = (_Float16)f.x;
      As[arow][acb + 4 * t + 1] = (_Float16)f.y;
      As[arow][acb + 4 * t + 2] = (_Float16)f.z;
      As[arow][acb + 4 * t + 3] = (_Float16)f.w;
    }
    // ---- stage B transposed: Bt[n][k] = W[k0+k][n_blk+n], paired-k stores --
    {
      const float* w0 = W + (size_t)(k0 + bk) * N + n_blk + bn;
      const float* w1 = w0 + N;
      float r0[8], r1[8];
#pragma unroll
      for (int t = 0; t < 2; ++t) {
        float4 f0 = reinterpret_cast<const float4*>(w0)[t];
        float4 f1 = reinterpret_cast<const float4*>(w1)[t];
        r0[4 * t + 0] = f0.x; r0[4 * t + 1] = f0.y; r0[4 * t + 2] = f0.z; r0[4 * t + 3] = f0.w;
        r1[4 * t + 0] = f1.x; r1[4 * t + 1] = f1.y; r1[4 * t + 2] = f1.z; r1[4 * t + 3] = f1.w;
      }
#pragma unroll
      for (int j = 0; j < 8; ++j) {
        h2 pr; pr[0] = (_Float16)r0[j]; pr[1] = (_Float16)r1[j];
        *reinterpret_cast<h2*>(&Bt[bn + j][bk]) = pr;
      }
      if (k0 + TK < K) __builtin_prefetch(w0 + (size_t)TK * N, 0, 3);
    }
    __syncthreads();
    v16h a0 = load_a_frag(&As[0][0], wm,      lane);
    v16h a1 = load_a_frag(&As[0][0], wm + 16, lane);
#pragma unroll
    for (int j = 0; j < 4; ++j) {
      v16h bf = load_b_frag_t(&Bt[0][0], wn + 16 * j, lane);
      acc[0][j] = wmma_f16(a0, bf, acc[0][j]);
      acc[1][j] = wmma_f16(a1, bf, acc[1][j]);
    }
    __syncthreads();
  }

  const int rbase = (lane >> 4) * 8;   // C/D: VGPR r -> row r (+8 upper half)
  const int col   = lane & 15;
#pragma unroll
  for (int i = 0; i < 2; ++i)
#pragma unroll
    for (int j = 0; j < 4; ++j)
#pragma unroll
      for (int r = 0; r < 8; ++r)
        Y[(size_t)(m_blk + wm + 16 * i + rbase + r) * N +
          (n_blk + wn + 16 * j + col)] = (_Float16)acc[i][j][r];
}

// -------------------- 2) masked score GEMM: q1 @ k1^T -----------------------
// B = k1^T, so Bt[n][k] = k1[n][k]: staging is a straight vector copy.
__global__ __launch_bounds__(256) void scores_gemm(
    const _Float16* __restrict__ Q1, const _Float16* __restrict__ K1,
    float* __restrict__ Sc) {
  const int b = blockIdx.z;
  const _Float16* Qb = Q1 + (size_t)b * SEQ * DIM;
  const _Float16* Kb = K1 + (size_t)b * SEQ * DIM;
  float* Sb = Sc + (size_t)b * SEQ * SEQ;
  const int tid   = threadIdx.x;
  const int m_blk = blockIdx.x * TM;   // query rows
  const int n_blk = blockIdx.y * TN;   // key cols

  // Fully-masked tile above the diagonal: write -inf, skip the GEMM.
  if (n_blk >= m_blk + TM) {
    for (int e = tid; e < TM * TN; e += 256) {
      int r = e >> 7, c = e & 127;
      Sb[(size_t)(m_blk + r) * SEQ + n_blk + c] = NEGINF;
    }
    return;
  }

  __shared__ alignas(16) _Float16 As[TM][LDA];
  __shared__ alignas(16) _Float16 Bt[TN][LDBT];
  const int lane = tid & 31;
  const int wave = tid >> 5;
  const int wm = (wave >> 1) * 32;
  const int wn = (wave & 1) * 64;
  const int arow = tid >> 1, acb = (tid & 1) * 16;
  const int brow = tid >> 1, bcb = (tid & 1) * 16;
  v8f acc[2][4] = {};

  for (int k0 = 0; k0 < DIM; k0 += TK) {
    const _Float16* qp = Qb + (size_t)(m_blk + arow) * DIM + k0 + acb;
    *reinterpret_cast<h8*>(&As[arow][acb])     = *reinterpret_cast<const h8*>(qp);
    *reinterpret_cast<h8*>(&As[arow][acb + 8]) = *reinterpret_cast<const h8*>(qp + 8);
    const _Float16* kp = Kb + (size_t)(n_blk + brow) * DIM + k0 + bcb;
    *reinterpret_cast<h8*>(&Bt[brow][bcb])     = *reinterpret_cast<const h8*>(kp);
    *reinterpret_cast<h8*>(&Bt[brow][bcb + 8]) = *reinterpret_cast<const h8*>(kp + 8);
    __syncthreads();
    v16h a0 = load_a_frag(&As[0][0], wm,      lane);
    v16h a1 = load_a_frag(&As[0][0], wm + 16, lane);
#pragma unroll
    for (int j = 0; j < 4; ++j) {
      v16h bf = load_b_frag_t(&Bt[0][0], wn + 16 * j, lane);
      acc[0][j] = wmma_f16(a0, bf, acc[0][j]);
      acc[1][j] = wmma_f16(a1, bf, acc[1][j]);
    }
    __syncthreads();
  }

  const int rbase = (lane >> 4) * 8;
  const int col   = lane & 15;
#pragma unroll
  for (int i = 0; i < 2; ++i)
#pragma unroll
    for (int j = 0; j < 4; ++j)
#pragma unroll
      for (int r = 0; r < 8; ++r) {
        const int gr = m_blk + wm + 16 * i + rbase + r;
        const int gc = n_blk + wn + 16 * j + col;
        const float val = acc[i][j][r] * SCALE;
        Sb[(size_t)gr * SEQ + gc] = (gc <= gr) ? val : NEGINF;
      }
}

// ------------------------- 3) row softmax -> f16 P --------------------------
__global__ __launch_bounds__(256) void softmax_rows(
    const float* __restrict__ Sc, _Float16* __restrict__ P) {
  const size_t row = blockIdx.x;           // 0 .. B*SEQ-1
  const float* src = Sc + row * SEQ;
  _Float16*    dst = P  + row * SEQ;
  const int tid = threadIdx.x, lane = tid & 31, wave = tid >> 5;
  __shared__ float red[8];
  float vals[8];
  float m = -3.4e38f;
#pragma unroll
  for (int t = 0; t < 8; ++t) { vals[t] = src[tid + 256 * t]; m = fmaxf(m, vals[t]); }
#pragma unroll
  for (int off = 16; off >= 1; off >>= 1) m = fmaxf(m, __shfl_xor(m, off));
  if (lane == 0) red[wave] = m;
  __syncthreads();
  if (tid == 0) {
    float x = red[0];
    for (int i = 1; i < 8; ++i) x = fmaxf(x, red[i]);
    red[0] = x;
  }
  __syncthreads();
  m = red[0];
  __syncthreads();
  float s = 0.f;
#pragma unroll
  for (int t = 0; t < 8; ++t) { vals[t] = __expf(vals[t] - m); s += vals[t]; }
#pragma unroll
  for (int off = 16; off >= 1; off >>= 1) s += __shfl_xor(s, off);
  if (lane == 0) red[wave] = s;
  __syncthreads();
  if (tid == 0) {
    float x = 0.f;
    for (int i = 0; i < 8; ++i) x += red[i];
    red[0] = x;
  }
  __syncthreads();
  const float inv = 1.0f / red[0];
#pragma unroll
  for (int t = 0; t < 8; ++t) dst[tid + 256 * t] = (_Float16)(vals[t] * inv);
}

// ------------------- 4) O = P @ v1, K truncated at diagonal -----------------
__global__ __launch_bounds__(256) void pv_gemm(
    const _Float16* __restrict__ P, const _Float16* __restrict__ V1,
    float* __restrict__ O) {
  __shared__ alignas(16) _Float16 As[TM][LDA];
  __shared__ alignas(16) _Float16 Bt[TN][LDBT];
  const int b = blockIdx.z;
  const _Float16* Pb = P  + (size_t)b * SEQ * SEQ;
  const _Float16* Vb = V1 + (size_t)b * SEQ * DIM;
  float* Ob = O + (size_t)b * SEQ * DIM;
  const int tid  = threadIdx.x;
  const int lane = tid & 31;
  const int wave = tid >> 5;
  const int m_blk = blockIdx.x * TM;
  const int n_blk = blockIdx.y * TN;
  const int wm = (wave >> 1) * 32;
  const int wn = (wave & 1) * 64;
  const int arow = tid >> 1, acb = (tid & 1) * 16;
  const int bk = (tid & 15) * 2;     // B: rows k, k+1
  const int bn = (tid >> 4) * 8;     //    8 cols each
  v8f acc[2][4] = {};

  const int kmax = (m_blk + TM < SEQ) ? (m_blk + TM) : SEQ;  // causal: P==0 past diag
  for (int k0 = 0; k0 < kmax; k0 += TK) {
    const _Float16* pp = Pb + (size_t)(m_blk + arow) * SEQ + k0 + acb;
    *reinterpret_cast<h8*>(&As[arow][acb])     = *reinterpret_cast<const h8*>(pp);
    *reinterpret_cast<h8*>(&As[arow][acb + 8]) = *reinterpret_cast<const h8*>(pp + 8);
    // Bt[n][k] = V1[k0+k][n_blk+n], paired-k f16x2 stores
    {
      const _Float16* v0 = Vb + (size_t)(k0 + bk) * DIM + n_blk + bn;
      h8 r0 = *reinterpret_cast<const h8*>(v0);
      h8 r1 = *reinterpret_cast<const h8*>(v0 + DIM);
#pragma unroll
      for (int j = 0; j < 8; ++j) {
        h2 pr; pr[0] = r0[j]; pr[1] = r1[j];
        *reinterpret_cast<h2*>(&Bt[bn + j][bk]) = pr;
      }
      if (k0 + TK < kmax) __builtin_prefetch(v0 + (size_t)TK * DIM, 0, 3);
    }
    __syncthreads();
    v16h a0 = load_a_frag(&As[0][0], wm,      lane);
    v16h a1 = load_a_frag(&As[0][0], wm + 16, lane);
#pragma unroll
    for (int j = 0; j < 4; ++j) {
      v16h bf = load_b_frag_t(&Bt[0][0], wn + 16 * j, lane);
      acc[0][j] = wmma_f16(a0, bf, acc[0][j]);
      acc[1][j] = wmma_f16(a1, bf, acc[1][j]);
    }
    __syncthreads();
  }

  const int rbase = (lane >> 4) * 8;
  const int col   = lane & 15;
#pragma unroll
  for (int i = 0; i < 2; ++i)
#pragma unroll
    for (int j = 0; j < 4; ++j)
#pragma unroll
      for (int r = 0; r < 8; ++r)
        Ob[(size_t)(m_blk + wm + 16 * i + rbase + r) * DIM +
           (n_blk + wn + 16 * j + col)] = acc[i][j][r];
}

// ----------------------------------------------------------------------------
extern "C" void kernel_launch(void* const* d_in, const int* in_sizes, int n_in,
                              void* d_out, int out_size, void* d_ws, size_t ws_size,
                              hipStream_t stream) {
  const float* q  = (const float*)d_in[0];
  const float* k  = (const float*)d_in[1];
  const float* v  = (const float*)d_in[2];
  // d_in[3] = mask: causal, applied analytically (gc <= gr), not read.
  const float* Wq = (const float*)d_in[4];
  const float* Wk = (const float*)d_in[5];
  const float* Wv = (const float*)d_in[6];
  float* out = (float*)d_out;

  // Workspace layout: q1|k1|v1 (f16, 16MB each) | Sc (f32, 64MB) | P (f16, 32MB)
  char* ws = (char*)d_ws;
  const size_t projBytes = (size_t)BATCH * SEQ * DIM * sizeof(_Float16);
  _Float16* q1 = (_Float16*)(ws);
  _Float16* k1 = (_Float16*)(ws + projBytes);
  _Float16* v1 = (_Float16*)(ws + 2 * projBytes);
  float*    Sc = (float*)   (ws + 3 * projBytes);
  _Float16* Pm = (_Float16*)(ws + 3 * projBytes +
                             (size_t)BATCH * SEQ * SEQ * sizeof(float));

  const dim3 blk(256);
  const dim3 gProj((BATCH * SEQ) / TM, DIM / TN);       // 64 x 8
  proj_gemm_f16<<<gProj, blk, 0, stream>>>(q, Wq, q1, BATCH * SEQ, DIM, DIM);
  proj_gemm_f16<<<gProj, blk, 0, stream>>>(k, Wk, k1, BATCH * SEQ, DIM, DIM);
  proj_gemm_f16<<<gProj, blk, 0, stream>>>(v, Wv, v1, BATCH * SEQ, DIM, DIM);

  const dim3 gSc(SEQ / TM, SEQ / TN, BATCH);            // 16 x 16 x 4
  scores_gemm<<<gSc, blk, 0, stream>>>(q1, k1, Sc);

  softmax_rows<<<dim3(BATCH * SEQ), blk, 0, stream>>>(Sc, Pm);

  const dim3 gPV(SEQ / TM, DIM / TN, BATCH);            // 16 x 8 x 4
  pv_gemm<<<gPV, blk, 0, stream>>>(Pm, v1, out);
}
